// NeuralCDE_58669253263676
// MI455X (gfx1250) — compile-verified, matrix operand order checked by reading, not run on hardware
//
#include <hip/hip_runtime.h>
#include <hip/hip_fp16.h>
#include <math.h>

typedef _Float16 f16;
typedef __attribute__((ext_vector_type(16))) _Float16 v16h;
typedef __attribute__((ext_vector_type(8)))  float    v8f;

#define NB    128   // batch
#define CINC  128   // input channels
#define HIDC  128   // hidden
#define NSTEP 63    // scan steps
#define NWG   16    // workgroups in scan kernel
#define HSL   8     // h-columns owned per workgroup (128 / 16)

// ---------- small device helpers ----------
__device__ __forceinline__ float d_tanh(float x) {
#if __has_builtin(__builtin_amdgcn_tanhf)
  return __builtin_amdgcn_tanhf(x);
#else
  return tanhf(x);
#endif
}
__device__ __forceinline__ float d_sigmoid(float x) { return 1.f / (1.f + __expf(-x)); }

// fragment swizzle: (row/col = l&15) and K pattern for 16-bit WMMA operands (wave32)
__device__ __forceinline__ int swz_e(int kk) {           // kk = k & 31 -> element 0..15
  return ((kk >> 4) << 3) | (((kk >> 1) & 3) << 1) | (kk & 1);
}
__device__ __forceinline__ int swz_l(int rc, int kk) {   // lane 0..31
  return (rc & 15) | (((kk >> 3) & 1) << 4);
}

__device__ __forceinline__ v8f wmma16(v16h a, v16h b, v8f c) {
  return __builtin_amdgcn_wmma_f32_16x16x32_f16(false, a, false, b, (short)0, c, false, false);
}

// ---------- prep kernels ----------

__global__ void k_zero(unsigned* bar) {
  if (threadIdx.x == 0 && blockIdx.x == 0) *bar = 0u;
}

// convert fp32 row-major (K,N) weight into f16 B-fragment layout:
// out[((nt*(K/32)+kf)*32 + lane)*16 + e]
__global__ void k_swz(const float* __restrict__ W, f16* __restrict__ out, int K, int N) {
  const int id = blockIdx.x * blockDim.x + threadIdx.x;
  if (id >= K * N) return;
  const int k = id / N, n = id % N;
  const int nt = n >> 4, kf = k >> 5, kk = k & 31;
  const int e  = swz_e(kk);
  const int lw = (n & 15) | (((kk >> 3) & 1) << 4);
  out[((nt * (K >> 5) + kf) * 32 + lw) * 16 + e] = (f16)W[id];
}

// m/s segments: s[t][b][c] = x[b,c,t+1]-x[b,c,t]; m[t]=s[max(t-1,0)]
__global__ void k_ms(const float* __restrict__ x, float* __restrict__ mb, float* __restrict__ sb) {
  const int id = blockIdx.x * blockDim.x + threadIdx.x;
  if (id >= NSTEP * NB * CINC) return;
  const int t = id / (NB * CINC), r = id % (NB * CINC);
  const int b = r >> 7, c = r & 127;
  const size_t base = ((size_t)b * CINC + c) * 64;
  const float sv = x[base + t + 1] - x[base + t];
  const int tm = (t == 0) ? 0 : t - 1;
  const float mv = x[base + tm + 1] - x[base + tm];
  sb[id] = sv;
  mb[id] = mv;
}

// time embedding MLPs + FiLM params + z0 (with GroupNorm-8)
__global__ void __launch_bounds__(256)
k_temb(const float* __restrict__ x, const float* __restrict__ timev,
       const float* __restrict__ te_w1, const float* __restrict__ te_b1,
       const float* __restrict__ te_w2, const float* __restrict__ te_b2,
       const float* __restrict__ fi_tw, const float* __restrict__ fi_tb,
       const float* __restrict__ fi_w,  const float* __restrict__ fi_b,
       const float* __restrict__ fi_gw, const float* __restrict__ fi_gb,
       const float* __restrict__ fl_tw, const float* __restrict__ fl_tb,
       float* __restrict__ tm2buf, float* __restrict__ zb0) {
  __shared__ float t0[64], h1a[256], stv[256], tmv[256];
  const int b = blockIdx.x, tid = threadIdx.x;
  if (tid < 64) {
    const float f = __expf(-logf(10000.f) * (float)(tid & 31) / 31.f);
    const float ang = timev[b] * f;
    t0[tid] = (tid < 32) ? sinf(ang) : cosf(ang);
  }
  __syncthreads();
  {
    float a = te_b1[tid];
    for (int k = 0; k < 64; ++k) a += t0[k] * te_w1[k * 256 + tid];
    h1a[tid] = 0.5f * a * (1.f + erff(a * 0.70710678118f));   // exact gelu
  }
  __syncthreads();
  {
    float a = te_b2[tid];
    for (int k = 0; k < 256; ++k) a += h1a[k] * te_w2[k * 256 + tid];
    stv[tid] = a * d_sigmoid(a);                              // silu(temb)
  }
  __syncthreads();
  {
    float a = fi_tb[tid];
    for (int k = 0; k < 256; ++k) a += stv[k] * fi_tw[k * 256 + tid];
    tmv[tid] = a;                                             // scale|shift
  }
  if (tid < 128) {
    float a = fl_tb[tid];
    for (int k = 0; k < 256; ++k) a += stv[k] * fl_tw[k * 128 + tid];
    tm2buf[b * 128 + tid] = a;                                // sc2|sh2 for final FiLM
  }
  __syncthreads();
  if (tid < 128) {
    float a = fi_b[tid];
    for (int c = 0; c < 128; ++c) a += x[((size_t)b * CINC + c) * 64] * fi_w[c * 128 + tid];
    float s = a;
    s += __shfl_xor(s, 1); s += __shfl_xor(s, 2); s += __shfl_xor(s, 4); s += __shfl_xor(s, 8);
    const float mu = s * (1.f / 16.f);
    const float d = a - mu;
    float q = d * d;
    q += __shfl_xor(q, 1); q += __shfl_xor(q, 2); q += __shfl_xor(q, 4); q += __shfl_xor(q, 8);
    const float var = q * (1.f / 16.f);
    float xn = d * rsqrtf(var + 1e-5f);
    xn = xn * fi_gw[tid] + fi_gb[tid];
    zb0[b * 128 + tid] = xn * (tmv[tid] + 1.f) + tmv[128 + tid];
  }
}

// ---------- the persistent CDE scan kernel (16 WGs x 512 threads) ----------

// one MLP layer: each wave owns a fixed M-tile (mt = wv>>1) so A-fragments are
// loaded once from LDS and reused across its 8 N-tiles. Weights stream from L2
// as B-fragments and are consumed immediately (no spill pressure).
__device__ __forceinline__ void mlp_layer(const f16* __restrict__ inb, int KF,
                                          const v16h* __restrict__ wsw,
                                          const float* __restrict__ bias,
                                          f16* __restrict__ outb, int wv, int lane) {
  const int mt = wv >> 1;
  v16h af[8];
  for (int kf = 0; kf < KF; ++kf)
    af[kf] = *(const v16h*)(inb + ((mt * KF + kf) * 32 + lane) * 16);
#pragma unroll 1
  for (int i = 0; i < 8; ++i) {
    const int nt = (wv & 1) * 8 + i;
    const float bv = bias[nt * 16 + (lane & 15)];
    v8f acc = {bv, bv, bv, bv, bv, bv, bv, bv};
#pragma unroll
    for (int kf = 0; kf < 8; ++kf) {
      if (kf < KF) {
        const v16h bfr = wsw[(nt * KF + kf) * 32 + lane];
        acc = wmma16(af[kf], bfr, acc);
      }
    }
    const int hi = lane >> 4;
    const int nn = nt * 16 + (lane & 15);
    const int kf2 = nn >> 5, kk = nn & 31;
    const int e = swz_e(kk);
#pragma unroll
    for (int r = 0; r < 8; ++r) {
      float v = acc[r]; v = v > 0.f ? v : 0.f;   // relu
      const int mm = mt * 16 + r + 8 * hi;
      outb[((mt * 8 + kf2) * 32 + swz_l(mm, kk)) * 16 + e] = (f16)v;
    }
  }
}

__global__ void __launch_bounds__(512)
k_scan(float* __restrict__ zb0, float* __restrict__ zb1, float* __restrict__ kb,
       const float* __restrict__ mb, const float* __restrict__ sb,
       const f16* __restrict__ w0s, const f16* __restrict__ w1s,
       const f16* __restrict__ w2s, const f16* __restrict__ w3s,
       const f16* __restrict__ wfs,
       const float* __restrict__ b0, const float* __restrict__ b1,
       const float* __restrict__ b2, const float* __restrict__ b3,
       const float* __restrict__ bf, unsigned* bar) {
  __shared__ __align__(64) unsigned char SMEM[160 * 1024];
  f16*   ldsZ  = (f16*)SMEM;                    // 32KB: z_eval A-frags [8][4][32][16]
  float* ldsK  = (float*)SMEM;                  // alias 32KB: k partials [128][HSL][8]
  f16*   ldsHa = (f16*)(SMEM + 32 * 1024);      // 64KB: [8][8][32][16]
  f16*   ldsHb = (f16*)(SMEM + 96 * 1024);      // 64KB
  const v16h* w0v = (const v16h*)w0s;
  const v16h* w1v = (const v16h*)w1s;
  const v16h* w2v = (const v16h*)w2s;
  const v16h* w3v = (const v16h*)w3s;
  const v16h* wfv = (const v16h*)wfs;

  const int tid = threadIdx.x;
  const int wv = tid >> 5, lane = tid & 31;
  const int wg = blockIdx.x;
  unsigned round = 0;

  for (int step = 0; step < NSTEP; ++step) {
    const float* zin  = (step & 1) ? zb1 : zb0;
    float*       zout = (step & 1) ? zb0 : zb1;
    for (int rk = 0; rk < 4; ++rk) {
      // ---- phase A0: build z_eval (RK4 stage input) as f16 A-fragments ----
      {
        const int b = tid >> 2, seg = tid & 3;    // 512 threads x 32 elems
        const int base = b * HIDC + seg * 32;
#pragma unroll 8
        for (int ii = 0; ii < 32; ++ii) {
          const int d = seg * 32 + ii;
          float z = zin[base + ii];
          if (rk == 1)      z += kb[base + ii] * (1.f / 3.f);
          else if (rk == 2) z += kb[16384 + base + ii] - kb[base + ii] * (1.f / 3.f);
          else if (rk == 3) z += kb[base + ii] - kb[16384 + base + ii] + kb[32768 + base + ii];
          const int kk = d & 31;
          ldsZ[(((b >> 4) * 4 + (d >> 5)) * 32 + swz_l(b, kk)) * 16 + swz_e(kk)] = (f16)z;
        }
      }
      __syncthreads();
      // ---- phase A: 4-layer MLP (redundant per WG, all 128 batch rows) ----
      mlp_layer(ldsZ, 4, w0v, b0, ldsHa, wv, lane); __syncthreads();
      mlp_layer(ldsHa, 8, w1v, b1, ldsHb, wv, lane); __syncthreads();
      mlp_layer(ldsHb, 8, w2v, b2, ldsHa, wv, lane); __syncthreads();
      mlp_layer(ldsHa, 8, w3v, b3, ldsHb, wv, lane); __syncthreads();
      // ---- phase B: F = tanh(H3 @ Wf_slice) fused with k += F * dX ----
      {
        const int m = wv >> 1;                    // fixed M-tile per wave
        v16h af[8];
#pragma unroll
        for (int kf = 0; kf < 8; ++kf)
          af[kf] = *(const v16h*)(ldsHb + ((m * 8 + kf) * 32 + lane) * 16);
        const int hi = lane >> 4;
#pragma unroll 1
        for (int i2 = 0; i2 < 4; ++i2) {
          const int ct = (wv & 1) * 4 + i2;
          const int cc = ct * 16 + (lane & 15);
          float dxv[8];
#pragma unroll
          for (int r = 0; r < 8; ++r) {
            const int b = 16 * m + r + 8 * hi;
            const size_t off = ((size_t)step * NB + b) * CINC + cc;
            const float mv = mb[off], sv = sb[off];
            // frac 0 -> m ; 1/3 -> s ; 2/3 -> m+4/3(s-m) ; 1 -> s
            dxv[r] = (rk == 0) ? mv : ((rk == 2) ? mv + (4.f / 3.f) * (sv - mv) : sv);
          }
#pragma unroll 1
          for (int h = 0; h < HSL; ++h) {
            const int hg = wg * HSL + h;
            const float bv = bf[hg * CINC + cc];
            v8f acc = {bv, bv, bv, bv, bv, bv, bv, bv};
            const int ntile = hg * 8 + ct;
#pragma unroll
            for (int kf = 0; kf < 8; ++kf) {
              const v16h bfr = wfv[(ntile * 8 + kf) * 32 + lane];
              acc = wmma16(af[kf], bfr, acc);
            }
#pragma unroll
            for (int r = 0; r < 8; ++r) {
              float p = d_tanh(acc[r]) * dxv[r];
              p += __shfl_xor(p, 1); p += __shfl_xor(p, 2);
              p += __shfl_xor(p, 4); p += __shfl_xor(p, 8);
              if ((lane & 15) == r) {             // lanes r and r+16 -> two b rows
                const int b = 16 * m + r + 8 * hi;
                ldsK[(b * HSL + h) * 8 + ct] = p; // unique slot: deterministic
              }
            }
          }
        }
      }
      __syncthreads();
      // ---- readout: finalize k slice, update z after rk3 ----
#pragma unroll
      for (int j = 0; j < 2; ++j) {
        const int idx = tid + 512 * j;            // 1024 (b,hl) entries
        const int b = idx >> 3, hl = idx & 7;
        const int hg = wg * HSL + hl;
        float kv = 0.f;
#pragma unroll
        for (int ct = 0; ct < 8; ++ct) kv += ldsK[(b * HSL + hl) * 8 + ct];
        kb[rk * 16384 + b * HIDC + hg] = kv;
        if (rk == 3) {
          const float k1 = kb[b * HIDC + hg];
          const float k2 = kb[16384 + b * HIDC + hg];
          const float k3 = kb[32768 + b * HIDC + hg];
          zout[b * HIDC + hg] = zin[b * HIDC + hg] + 0.125f * (k1 + 3.f * (k2 + k3) + kv);
        }
      }
      // ---- grid barrier across the 16 workgroups ----
      ++round;
      __threadfence();
      __syncthreads();
      if (tid == 0) {
        atomicAdd(bar, 1u);
        const unsigned tgt = (unsigned)NWG * round;
        while (atomicAdd(bar, 0u) < tgt) __builtin_amdgcn_s_sleep(2);
      }
      __syncthreads();
      __threadfence();
    }
  }
}

// ---------- decoder / head kernels ----------

__global__ void k_up1(const float* __restrict__ zT, const float* __restrict__ w,
                      const float* __restrict__ bias, float* __restrict__ out) {
  const int id = blockIdx.x * blockDim.x + threadIdx.x;   // (b,o,p) = 128*32*256
  if (id >= 128 * 32 * 256) return;
  const int p = id & 255, o = (id >> 8) & 31, b = id >> 13;
  float a = bias[o];
#pragma unroll
  for (int t = 0; t < 3; ++t) {
    const int q = p + t - 1;
    if (q >= 0 && q < 256) a += w[o * 3 + t] * zT[b * 128 + (q >> 1)];
  }
  out[id] = a;
}

__global__ void __launch_bounds__(256)
k_up2(const float* __restrict__ h1, const float* __restrict__ w,
      const float* __restrict__ bias, float* __restrict__ out) {
  __shared__ float hs[32 * 256];
  __shared__ float wsr[96];
  const int bo = blockIdx.x;                // (b,o) = 128*64
  const int b = bo >> 6, o = bo & 63;
  const int tid = threadIdx.x;
  for (int j = tid; j < 32 * 256; j += 256) hs[j] = h1[b * 32 * 256 + j];
  if (tid < 96) wsr[tid] = w[o * 96 + tid];
  __syncthreads();
  const float bb = bias[o];
  for (int p = tid; p < 512; p += 256) {
    float a = bb;
    for (int c = 0; c < 32; ++c) {
#pragma unroll
      for (int t = 0; t < 3; ++t) {
        const int q = p + t - 1;
        if (q >= 0 && q < 512) a += wsr[c * 3 + t] * hs[c * 256 + (q >> 1)];
      }
    }
    out[(b * 64 + o) * 512 + p] = a;
  }
}

// fused conv3 (up3) + final matmul with fl_w  ->  ybuf[b][co][o]
__global__ void __launch_bounds__(1024)
k_fl(const float* __restrict__ h2, const float* __restrict__ w3,
     const float* __restrict__ b3, const float* __restrict__ flw,
     const float* __restrict__ flb, float* __restrict__ ybuf) {
  __shared__ float hs[64 * 512];   // 128KB
  __shared__ float row[1024];
  __shared__ float wrow[192];
  __shared__ float part[1024];
  const int b = blockIdx.x, tid = threadIdx.x;
  for (int j = tid; j < 64 * 512; j += 1024) hs[j] = h2[(size_t)b * 64 * 512 + j];
  __syncthreads();
  for (int co = 0; co < 128; ++co) {
    if (tid < 192) wrow[tid] = w3[co * 192 + tid];
    __syncthreads();
    {
      const int p = tid;
      float a = b3[co];
      for (int c = 0; c < 64; ++c) {
#pragma unroll
        for (int t = 0; t < 3; ++t) {
          const int q = p + t - 1;
          if (q >= 0 && q < 1024) a += wrow[c * 3 + t] * hs[c * 512 + (q >> 1)];
        }
      }
      row[p] = a;
    }
    __syncthreads();
    {
      const int o = tid & 63, ch = tid >> 6;
      float a = 0.f;
      const int p0 = ch * 64;
      for (int j = 0; j < 64; ++j) a += row[p0 + j] * flw[(p0 + j) * 64 + o];
      part[tid] = a;
    }
    __syncthreads();
    if (tid < 64) {
      float a = flb[tid];
      for (int ch = 0; ch < 16; ++ch) a += part[ch * 64 + tid];
      ybuf[((size_t)b * 128 + co) * 64 + tid] = a;
    }
    __syncthreads();
  }
}

// GroupNorm(2) over 64 channels + FiLM + silu -> out (B,64,128)
__global__ void __launch_bounds__(256)
k_gn(const float* __restrict__ ybuf, const float* __restrict__ gw,
     const float* __restrict__ gb, const float* __restrict__ tm2,
     float* __restrict__ out) {
  __shared__ float rs[256], rq[256];
  const int bg = blockIdx.x;
  const int b = bg >> 1, g = bg & 1;
  const int tid = threadIdx.x;
  float s = 0.f, q = 0.f;
  for (int j = 0; j < 16; ++j) {
    const int idx = tid + j * 256;             // 4096 elems/group
    const int o = g * 32 + (idx >> 7), c = idx & 127;
    const float v = ybuf[((size_t)b * 128 + c) * 64 + o];
    s += v; q += v * v;
  }
  rs[tid] = s; rq[tid] = q;
  __syncthreads();
  for (int off = 128; off > 0; off >>= 1) {
    if (tid < off) { rs[tid] += rs[tid + off]; rq[tid] += rq[tid + off]; }
    __syncthreads();
  }
  const float mu  = rs[0] * (1.f / 4096.f);
  const float var = rq[0] * (1.f / 4096.f) - mu * mu;
  const float inv = rsqrtf(var + 1e-5f);
  for (int j = 0; j < 16; ++j) {
    const int idx = tid + j * 256;
    const int o = g * 32 + (idx >> 7), c = idx & 127;
    const float v = ybuf[((size_t)b * 128 + c) * 64 + o];
    float xn = (v - mu) * inv;
    xn = xn * gw[o] + gb[o];
    const float sc = tm2[b * 128 + o], sh = tm2[b * 128 + 64 + o];
    const float y = xn * (sc + 1.f) + sh;
    out[((size_t)b * 64 + o) * 128 + c] = y * d_sigmoid(y);
  }
}

// ---------- host launcher ----------
extern "C" void kernel_launch(void* const* d_in, const int* in_sizes, int n_in,
                              void* d_out, int out_size, void* d_ws, size_t ws_size,
                              hipStream_t stream) {
  (void)in_sizes; (void)n_in; (void)out_size; (void)ws_size;
  const float* x      = (const float*)d_in[0];
  const float* timev  = (const float*)d_in[1];
  const float* te_w1  = (const float*)d_in[2];
  const float* te_b1  = (const float*)d_in[3];
  const float* te_w2  = (const float*)d_in[4];
  const float* te_b2  = (const float*)d_in[5];
  const float* fi_tw  = (const float*)d_in[6];
  const float* fi_tb  = (const float*)d_in[7];
  const float* fi_w   = (const float*)d_in[8];
  const float* fi_b   = (const float*)d_in[9];
  const float* fi_gw  = (const float*)d_in[10];
  const float* fi_gb  = (const float*)d_in[11];
  const float* cde_w0 = (const float*)d_in[12];
  const float* cde_b0 = (const float*)d_in[13];
  const float* cde_w1 = (const float*)d_in[14];
  const float* cde_b1 = (const float*)d_in[15];
  const float* cde_w2 = (const float*)d_in[16];
  const float* cde_b2 = (const float*)d_in[17];
  const float* cde_w3 = (const float*)d_in[18];
  const float* cde_b3 = (const float*)d_in[19];
  const float* cde_wf = (const float*)d_in[20];
  const float* cde_bf = (const float*)d_in[21];
  const float* up1_w  = (const float*)d_in[22];
  const float* up1_b  = (const float*)d_in[23];
  const float* up2_w  = (const float*)d_in[24];
  const float* up2_b  = (const float*)d_in[25];
  const float* up3_w  = (const float*)d_in[26];
  const float* up3_b  = (const float*)d_in[27];
  const float* fl_tw  = (const float*)d_in[28];
  const float* fl_tb  = (const float*)d_in[29];
  const float* fl_w   = (const float*)d_in[30];
  const float* fl_b   = (const float*)d_in[31];
  const float* fl_gw  = (const float*)d_in[32];
  const float* fl_gb  = (const float*)d_in[33];

  char* ws = (char*)d_ws;
  // persistent small region
  unsigned* bar = (unsigned*)(ws + 0);
  float* zb0 = (float*)(ws + 256);
  float* zb1 = (float*)(ws + 256 + 65536);
  float* kb  = (float*)(ws + 256 + 2 * 65536);                 // 4 * 64KB
  float* tm2 = (float*)(ws + 256 + 2 * 65536 + 262144);
  const size_t OFF_W0 = 256 + 2 * 65536 + 262144 + 65536;      // 459008
  f16* w0s = (f16*)(ws + OFF_W0);
  f16* w1s = (f16*)(ws + OFF_W0 + 65536);
  f16* w2s = (f16*)(ws + OFF_W0 + 65536 + 131072);
  f16* w3s = (f16*)(ws + OFF_W0 + 65536 + 2 * 131072);
  f16* wfs = (f16*)(ws + OFF_W0 + 65536 + 3 * 131072);         // 8.39MB
  float* mb = (float*)(ws + OFF_W0 + 65536 + 3 * 131072 + 8388608);
  float* sb = (float*)((char*)mb + 4128768);
  // post-scan buffers alias the (dead) weight/segment region
  float* h1 = (float*)(ws + OFF_W0);                 // 4.19MB
  float* h2 = (float*)(ws + OFF_W0 + 4194304);       // 16.78MB
  float* yb = (float*)(ws + OFF_W0 + 4194304 + 16777216);

  k_zero<<<1, 64, 0, stream>>>(bar);
  k_swz<<<(128 * 256 + 255) / 256, 256, 0, stream>>>(cde_w0, w0s, 128, 256);
  k_swz<<<(256 * 256 + 255) / 256, 256, 0, stream>>>(cde_w1, w1s, 256, 256);
  k_swz<<<(256 * 256 + 255) / 256, 256, 0, stream>>>(cde_w2, w2s, 256, 256);
  k_swz<<<(256 * 256 + 255) / 256, 256, 0, stream>>>(cde_w3, w3s, 256, 256);
  k_swz<<<(256 * 16384 + 255) / 256, 256, 0, stream>>>(cde_wf, wfs, 256, 16384);
  k_ms<<<(NSTEP * NB * CINC + 255) / 256, 256, 0, stream>>>(x, mb, sb);
  k_temb<<<NB, 256, 0, stream>>>(x, timev, te_w1, te_b1, te_w2, te_b2,
                                 fi_tw, fi_tb, fi_w, fi_b, fi_gw, fi_gb,
                                 fl_tw, fl_tb, tm2, zb0);
  k_scan<<<NWG, 512, 0, stream>>>(zb0, zb1, kb, mb, sb,
                                  w0s, w1s, w2s, w3s, wfs,
                                  cde_b0, cde_b1, cde_b2, cde_b3, cde_bf, bar);
  // 63 steps (odd) starting from zb0 -> final zT lives in zb1
  k_up1<<<(128 * 32 * 256) / 256, 256, 0, stream>>>(zb1, up1_w, up1_b, h1);
  k_up2<<<128 * 64, 256, 0, stream>>>(h1, up2_w, up2_b, h2);
  k_fl<<<128, 1024, 0, stream>>>(h2, up3_w, up3_b, fl_w, fl_b, yb);
  k_gn<<<256, 256, 0, stream>>>(yb, fl_gw, fl_gb, tm2, (float*)d_out);
}
